// PrototypeAttentionLoss_62088047231356
// MI455X (gfx1250) — compile-verified, matrix operand order checked by reading, not run on hardware
//
#include <hip/hip_runtime.h>
#include <math.h>

// ---------------- problem constants ----------------
#define NB     4096      // batch B
#define BN     16        // bands per sample
#define DD     512       // feature dim
#define NC     9         // classes
#define ROWS   (NB*NC)   // 36864 = GEMM M dim
#define SCALE  0.125f    // (512/8)^-0.5
// ws byte offsets
#define QR_OFF    0u
#define QI_OFF    18432u
#define AR_OFF    36864u
#define AI_OFF    55296u
#define T_OFF     73728u
#define G_OFF     73764u
#define D2_OFF    73792u     // 36864 floats
#define DIST_OFF  221248u    // 36864 floats
#define PB_OFF    368704u    // 786432 halves (3 mats x 32 ntile x 16 kb x 512)

typedef __attribute__((ext_vector_type(16))) _Float16 v16h;
typedef __attribute__((ext_vector_type(8)))  _Float16 v8h;
typedef __attribute__((ext_vector_type(8)))  float    v8f;
typedef __attribute__((ext_vector_type(4)))  float    v4f;

static __device__ __forceinline__ v8f wmma16(v16h a, v16h b, v8f c) {
  return __builtin_amdgcn_wmma_f32_16x16x32_f16(false, a, false, b, (short)0, c, false, false);
}

// k-local index inside a 32-wide K block for 16-bit A/B fragments (ISA 7.12.2)
static __device__ __forceinline__ int klocal(int lane, int hh) {
  return (hh & 7) + ((hh >> 3) << 4) + (((lane >> 4) & 1) << 3);
}

// ---------------- kernel 0: zero accumulators ----------------
__global__ void k_zero(float* ws_zero, float* out) {
  int idx = blockIdx.x * blockDim.x + threadIdx.x;
  if (idx < ROWS * 2) ws_zero[idx] = 0.0f;   // D2 + Dist contiguous
  if (idx == 0) out[0] = 0.0f;
}

// ---------------- kernel 1: pack Wv panels to f16 B-fragment layout ----------------
// layout: [mat(3)][ntile(32)][kb(16)][lane(32)][hh(16)]  element = W[n, k] (== Wv^T[k,n])
__global__ void k_packB(const float* Wv_r, const float* Wv_i, _Float16* pB) {
  int idx = blockIdx.x * blockDim.x + threadIdx.x;          // < 786432
  int mat = idx >> 18;
  int r   = idx & 262143;
  int ntile = r >> 13;
  int r2  = r & 8191;
  int kb  = r2 >> 9;
  int r3  = r2 & 511;
  int lane = r3 >> 4;
  int hh   = r3 & 15;
  int n = ntile * 16 + (lane & 15);
  int k = kb * 32 + klocal(lane, hh);
  float v;
  if (mat == 0)      v = Wv_r[n * DD + k];
  else if (mat == 1) v = Wv_i[n * DD + k];
  else               v = -Wv_i[n * DD + k];
  pB[idx] = (_Float16)v;
}

// ---------------- kernel 2: q = cvlinear(prototypes) ----------------
__global__ void k_q(const float* proto, const float* Wq_r, const float* Wq_i,
                    const float* bq_r, const float* bq_i, float* qr, float* qi) {
  int idx = blockIdx.x * blockDim.x + threadIdx.x;          // < 9216
  int part = idx / (NC * DD);
  int r = idx % (NC * DD);
  int c = r / DD, d = r % DD;
  const float* pr = proto + c * DD;
  const float* pi = proto + NC * DD + c * DD;
  const float* wr = Wq_r + d * DD;
  const float* wi = Wq_i + d * DD;
  float s;
  if (part == 0) {                 // q_r = pr@Wr^T - pi@Wi^T + bq_r
    s = bq_r[d];
    for (int e = 0; e < DD; ++e) s += pr[e] * wr[e] - pi[e] * wi[e];
    qr[c * DD + d] = s;
  } else {                         // q_i = pr@Wi^T + pi@Wr^T + bq_i
    s = bq_i[d];
    for (int e = 0; e < DD; ++e) s += pr[e] * wi[e] + pi[e] * wr[e];
    qi[c * DD + d] = s;
  }
}

// ---------------- kernel 3: A = q folded into Wk, plus t[c] and softplus(gain) ----------------
__global__ void k_A(const float* qr, const float* qi,
                    const float* Wk_r, const float* Wk_i,
                    const float* bk_r, const float* bk_i,
                    const float* gin,
                    float* Ar, float* Ai, float* tvec, float* gsp) {
  int idx = blockIdx.x * blockDim.x + threadIdx.x;
  if (idx < 2 * NC * DD) {
    int part = idx / (NC * DD);
    int r = idx % (NC * DD);
    int c = r / DD, e = r % DD;
    const float* qrc = qr + c * DD;
    const float* qic = qi + c * DD;
    float s = 0.0f;
    if (part == 0) {   // A_r[c,e] = sum_d qr*Wk_r[d,e] + qi*Wk_i[d,e]
      for (int d = 0; d < DD; ++d) s += qrc[d] * Wk_r[d * DD + e] + qic[d] * Wk_i[d * DD + e];
      Ar[c * DD + e] = s;
    } else {           // A_i[c,e] = sum_d qi*Wk_r[d,e] - qr*Wk_i[d,e]
      for (int d = 0; d < DD; ++d) s += qic[d] * Wk_r[d * DD + e] - qrc[d] * Wk_i[d * DD + e];
      Ai[c * DD + e] = s;
    }
  } else if (idx < 2 * NC * DD + NC) {
    int c = idx - 2 * NC * DD;
    float s = 0.0f;
    for (int d = 0; d < DD; ++d) s += qr[c * DD + d] * bk_r[d] + qi[c * DD + d] * bk_i[d];
    tvec[c] = s;
  } else if (idx == 2 * NC * DD + NC) {
    gsp[0] = log1pf(expf(gin[0]));   // softplus(attn_gain)
  }
}

// ---------------- kernel 4: scores -> normalized -> softmax attn (one WG per sample) ----------------
__global__ void k_attn(const float* xr, const float* xi,
                       const float* Ar, const float* Ai,
                       const float* tvec, const float* gsp,
                       float* attn_out) {
  int b = blockIdx.x;
  int tid = threadIdx.x, wave = tid >> 5, lane = tid & 31;
  __shared__ float sS[NC * BN];
  const float* xrb = xr + (size_t)b * BN * DD;
  const float* xib = xi + (size_t)b * BN * DD;
  for (int e = wave; e < NC * BN; e += 8) {
    int c = e >> 4, n = e & 15;
    const v4f* A4  = (const v4f*)(Ar + c * DD);
    const v4f* Ai4 = (const v4f*)(Ai + c * DD);
    const v4f* Xr4 = (const v4f*)(xrb + n * DD);
    const v4f* Xi4 = (const v4f*)(xib + n * DD);
    float s = 0.0f;
#pragma unroll
    for (int q = 0; q < 4; ++q) {           // 128 v4f chunks striped across 32 lanes
      int p = lane + q * 32;
      v4f a = A4[p], x = Xr4[p];
      s += a.x * x.x + a.y * x.y + a.z * x.z + a.w * x.w;
      v4f ai = Ai4[p], y = Xi4[p];
      s += ai.x * y.x + ai.y * y.y + ai.z * y.z + ai.w * y.w;
    }
    for (int m = 16; m; m >>= 1) s += __shfl_xor(s, m, 32);
    if (lane == 0) sS[e] = (s + tvec[c]) * SCALE;
  }
  __syncthreads();
  if (tid < NC) {
    int c = tid;
    float g = gsp[0];
    float mean = 0.0f;
    for (int n = 0; n < BN; ++n) mean += sS[c * BN + n];
    mean *= (1.0f / BN);
    float var = 0.0f;
    for (int n = 0; n < BN; ++n) { float d = sS[c * BN + n] - mean; var += d * d; }
    float stdv = sqrtf(var / (BN - 1)) + 1e-6f;     // ddof=1
    float z[BN], mz = -1e30f;
    for (int n = 0; n < BN; ++n) { z[n] = (sS[c * BN + n] - mean) / stdv * g; mz = fmaxf(mz, z[n]); }
    float sum = 0.0f;
    for (int n = 0; n < BN; ++n) { z[n] = expf(z[n] - mz); sum += z[n]; }
    float* o = attn_out + ((size_t)b * NC + c) * BN;
    float inv = 1.0f / sum;
    for (int n = 0; n < BN; ++n) o[n] = z[n] * inv;
  }
}

// ---------------- kernel 5: WMMA GEMM hat = (attn@x)@Wv^T + bv, fused distance epilogue ----------------
// grid: ROWS/16 = 2304 WGs, 256 threads (8 waves). Each WG: 16 rows x full N=512.
__global__ void __launch_bounds__(256)
k_gemm(const float* xr, const float* xi, const float* attn,
       const _Float16* pB, const float* bv_r, const float* bv_i,
       const float* proto, float* D2, float* Dist) {
  __shared__ _Float16 shA[2 * 16 * 32 * 16];   // [mat][kb][lane][hh], 32 KB
  int row0 = blockIdx.x * 16;
  int tid = threadIdx.x, wave = tid >> 5, lane = tid & 31;

  // ---- stage A fragments: u[m,k] = sum_n attn[row,n] * x[b,n,k], f16, fragment order ----
  // thread owns (mat, m) and a contiguous 64-k run: vectorized loads + b128 LDS stores.
  {
    int sub  = tid & 7;          // which 64-k run
    int pair = tid >> 3;         // 0..31
    int mat  = pair >> 4;        // 0 = real, 1 = imag
    int m    = pair & 15;
    int k0   = sub * 64;
    int row  = row0 + m;
    int bidx = row / NC;
    // hoist attn row (16 floats, 64B aligned)
    const v4f* ap4 = (const v4f*)(attn + (size_t)row * BN);
    v4f a0 = ap4[0], a1 = ap4[1], a2 = ap4[2], a3 = ap4[3];
    float aw[16] = {a0.x,a0.y,a0.z,a0.w, a1.x,a1.y,a1.z,a1.w,
                    a2.x,a2.y,a2.z,a2.w, a3.x,a3.y,a3.z,a3.w};
    const float* xbase = (mat ? xi : xr) + (size_t)bidx * BN * DD + k0;
    v4f acc4[16];
#pragma unroll
    for (int q = 0; q < 16; ++q) acc4[q] = (v4f){0.f, 0.f, 0.f, 0.f};
#pragma unroll
    for (int n = 0; n < BN; ++n) {
      const v4f* xp = (const v4f*)(xbase + n * DD);
      float an = aw[n];
#pragma unroll
      for (int q = 0; q < 16; ++q) {
        v4f v = xp[q];
        acc4[q].x += an * v.x; acc4[q].y += an * v.y;
        acc4[q].z += an * v.z; acc4[q].w += an * v.w;
      }
    }
    // write 8 groups of 8 consecutive k as one b128 LDS store each
#pragma unroll
    for (int g = 0; g < 8; ++g) {
      int kk = k0 + g * 8;
      int kb = kk >> 5;
      int lane_t = m + (((kk >> 3) & 1) << 4);
      int hh0 = ((kk >> 4) & 1) << 3;
      v4f lo = acc4[g * 2], hi = acc4[g * 2 + 1];
      v8h h;
      h[0] = (_Float16)lo.x; h[1] = (_Float16)lo.y;
      h[2] = (_Float16)lo.z; h[3] = (_Float16)lo.w;
      h[4] = (_Float16)hi.x; h[5] = (_Float16)hi.y;
      h[6] = (_Float16)hi.z; h[7] = (_Float16)hi.w;
      *reinterpret_cast<v8h*>(&shA[mat * 8192 + kb * 512 + lane_t * 16 + hh0]) = h;
    }
  }
  __syncthreads();

  // ---- WMMA main loop ----
  v8f accR[4], accI[4];
  v8f zero = {0.f, 0.f, 0.f, 0.f, 0.f, 0.f, 0.f, 0.f};
#pragma unroll
  for (int t = 0; t < 4; ++t) { accR[t] = zero; accI[t] = zero; }

  for (int kb = 0; kb < 16; ++kb) {
    v16h ar = *reinterpret_cast<const v16h*>(&shA[kb * 512 + lane * 16]);
    v16h ai = *reinterpret_cast<const v16h*>(&shA[8192 + kb * 512 + lane * 16]);
#pragma unroll
    for (int t = 0; t < 4; ++t) {
      int ntile = wave * 4 + t;
      size_t base = ((size_t)ntile * 16 + kb) * 512 + lane * 16;
      v16h bwr  = *reinterpret_cast<const v16h*>(pB + base);                    // Wr
      v16h bwi  = *reinterpret_cast<const v16h*>(pB + 262144 + base);           // Wi
      v16h bwin = *reinterpret_cast<const v16h*>(pB + 524288 + base);           // -Wi
      accR[t] = wmma16(ar, bwr,  accR[t]);   // u_r @ Wr^T
      accR[t] = wmma16(ai, bwin, accR[t]);   // - u_i @ Wi^T
      accI[t] = wmma16(ar, bwi,  accI[t]);   // u_r @ Wi^T
      accI[t] = wmma16(ai, bwr,  accI[t]);   // + u_i @ Wr^T
    }
  }

  // ---- epilogue: add bias, distance contributions, reduce over N ----
  float sumD2[8], sumDist[8];
#pragma unroll
  for (int j = 0; j < 8; ++j) { sumD2[j] = 0.0f; sumDist[j] = 0.0f; }
  int moff = ((lane >> 4) & 1) << 3;      // lanes>=16 hold rows m=8..15
#pragma unroll
  for (int t = 0; t < 4; ++t) {
    int n = (wave * 4 + t) * 16 + (lane & 15);
    float bvr = bv_r[n], bvi = bv_i[n];
#pragma unroll
    for (int j = 0; j < 8; ++j) {
      int row = row0 + j + moff;
      int c = row % NC;
      float hr = accR[t][j] + bvr;
      float hi = accI[t][j] + bvi;
      float pr = proto[c * DD + n];
      float pi = proto[NC * DD + c * DD + n];
      float dr = hr - pr, di = hi - pi;
      sumD2[j] += dr * dr + di * di;
      float cross = fabsf(hr * pi - hi * pr);
      float magz = sqrtf(hr * hr + hi * hi) + 1e-8f;
      float orth = cross / magz;
      float dot = hr * pr + hi * pi;
      float sgn = (dot < 0.0f) ? (2.0f * magz - orth) : orth;
      float pmag = sqrtf(pr * pr + pi * pi);
      sumDist[j] += sgn + fabsf(magz - pmag);
    }
  }
#pragma unroll
  for (int j = 0; j < 8; ++j) {
    float a = sumD2[j], d = sumDist[j];
    for (int m = 8; m; m >>= 1) { a += __shfl_xor(a, m, 32); d += __shfl_xor(d, m, 32); }
    if ((lane & 15) == 0) {
      int row = row0 + j + moff;
      atomicAdd(&D2[row], a);
      atomicAdd(&Dist[row], d);
    }
  }
}

// ---------------- kernel 6: per-sample losses, argmax, mean reduce ----------------
__global__ void k_final(const float* D2, const float* Dist, const int* labels, float* out) {
  int b = blockIdx.x * blockDim.x + threadIdx.x;
  if (b >= NB) return;
  float d2[NC], di[NC];
  for (int c = 0; c < NC; ++c) { d2[c] = D2[b * NC + c]; di[c] = Dist[b * NC + c]; }
  // soft-MSE: r = softmax(-D2), sum r*D2
  float mx = -1e30f;
  for (int c = 0; c < NC; ++c) mx = fmaxf(mx, -d2[c]);
  float sum = 0.0f, acc = 0.0f;
  for (int c = 0; c < NC; ++c) { float e = expf(-d2[c] - mx); sum += e; acc += e * d2[c]; }
  float softmse = acc / sum;
  // classification: logits = -Dist, log-softmax
  float ml = -1e30f; int arg = 0;
  for (int c = 0; c < NC; ++c) { float l = -di[c]; if (l > ml) { ml = l; arg = c; } }
  float se = 0.0f;
  for (int c = 0; c < NC; ++c) se += expf(-di[c] - ml);
  float lse = ml + logf(se);
  float lcls = -(-di[labels[b]] - lse);
  atomicAdd(out, (lcls + 0.01f * softmse) * (1.0f / NB));
  out[1 + b] = (float)arg;   // preds
}

// ---------------- host launcher ----------------
extern "C" void kernel_launch(void* const* d_in, const int* in_sizes, int n_in,
                              void* d_out, int out_size, void* d_ws, size_t ws_size,
                              hipStream_t stream) {
  const float* band_r = (const float*)d_in[0];
  const float* band_i = (const float*)d_in[1];
  const int*   labels = (const int*)d_in[2];
  const float* proto  = (const float*)d_in[3];
  const float* Wq_r = (const float*)d_in[4];
  const float* Wq_i = (const float*)d_in[5];
  const float* bq_r = (const float*)d_in[6];
  const float* bq_i = (const float*)d_in[7];
  const float* Wk_r = (const float*)d_in[8];
  const float* Wk_i = (const float*)d_in[9];
  const float* bk_r = (const float*)d_in[10];
  const float* bk_i = (const float*)d_in[11];
  const float* Wv_r = (const float*)d_in[12];
  const float* Wv_i = (const float*)d_in[13];
  const float* bv_r = (const float*)d_in[14];
  const float* bv_i = (const float*)d_in[15];
  const float* gain = (const float*)d_in[16];

  char* ws = (char*)d_ws;
  float* qr   = (float*)(ws + QR_OFF);
  float* qi   = (float*)(ws + QI_OFF);
  float* Ar   = (float*)(ws + AR_OFF);
  float* Ai   = (float*)(ws + AI_OFF);
  float* tvec = (float*)(ws + T_OFF);
  float* gsp  = (float*)(ws + G_OFF);
  float* D2   = (float*)(ws + D2_OFF);
  float* Dist = (float*)(ws + DIST_OFF);
  _Float16* pB = (_Float16*)(ws + PB_OFF);

  float* out = (float*)d_out;
  float* attn_out = out + 1 + NB;   // (loss, preds[B], attn[B*C*Bn])

  k_zero <<<(ROWS * 2 + 255) / 256, 256, 0, stream>>>(D2, out);   // D2,Dist contiguous
  k_packB<<<786432 / 256, 256, 0, stream>>>(Wv_r, Wv_i, pB);
  k_q    <<<(2 * NC * DD) / 256, 256, 0, stream>>>(proto, Wq_r, Wq_i, bq_r, bq_i, qr, qi);
  k_A    <<<(2 * NC * DD + NC + 1 + 255) / 256, 256, 0, stream>>>(qr, qi, Wk_r, Wk_i, bk_r, bk_i,
                                                                  gain, Ar, Ai, tvec, gsp);
  k_attn <<<NB, 256, 0, stream>>>(band_r, band_i, Ar, Ai, tvec, gsp, attn_out);
  k_gemm <<<ROWS / 16, 256, 0, stream>>>(band_r, band_i, attn_out, pB, bv_r, bv_i,
                                         proto, D2, Dist);
  k_final<<<(NB + 255) / 256, 256, 0, stream>>>(D2, Dist, labels, out);
}
